// STLSTM_28372553957712
// MI455X (gfx1250) — matrix operand order repaired and checked by reading, not compiled, not run on hardware
//
#include <hip/hip_runtime.h>

#define B_  32
#define F_  10
#define NB_ 20
#define H_  128
#define G_  5
#define R_  2

typedef __bf16 bf16x16 __attribute__((ext_vector_type(16)));
typedef float  f32x8   __attribute__((ext_vector_type(8)));

#define ABST 136   // padded A-buffer row stride (bf16 elems): 16B-aligned rows, conflict-breaking
#define WST  136   // padded transposed weight-stage row stride (bf16 elems)

__device__ __forceinline__ unsigned short f2bf(float f) {
  unsigned int u = __float_as_uint(f);
  u += 0x7FFFu + ((u >> 16) & 1u);      // round-to-nearest-even
  return (unsigned short)(u >> 16);
}
__device__ __forceinline__ float sigm (float x) { return 1.f / (1.f + __expf(-x)); }
__device__ __forceinline__ float tanhf_(float x) { return 2.f / (1.f + __expf(-2.f * x)) - 1.f; }

union AB32 { uint4 q[2]; bf16x16 v; };

// A-operand fragment per ISA 16-bit A 16x32 layout:
// lanes 0-15: M=lane,    v0-3 = K kb..kb+7 (kb=kc*32),  v4-7 = K kb+16..kb+23
// lanes16-31: M=lane-16, same with kb = kc*32 + 8
__device__ __forceinline__ bf16x16 load_a_frag(const unsigned short* Ab, int m, int kc, int lane) {
  int row = m * 16 + (lane & 15);
  int kb  = kc * 32 + ((lane >> 4) & 1) * 8;
  const unsigned short* p = Ab + row * ABST + kb;
  AB32 u;
  u.q[0] = *(const uint4*)(p);
  u.q[1] = *(const uint4*)(p + 16);
  return u.v;
}

// B-operand fragment per ISA 16-bit B 32x16 layout:
// lanes 0-15: N=lane, K = kc*32 + 0..7,16..23 ; lanes 16-31: N=lane-16, K = kc*32 + 8..15,24..31
// Wstg is stored transposed [n][k] so each lane reads two contiguous 16-byte runs of k.
__device__ __forceinline__ bf16x16 load_b_frag(const unsigned short* Wstg, int nloc, int kc, int lane) {
  int col = nloc * 16 + (lane & 15);
  int kb  = kc * 32 + ((lane >> 4) & 1) * 8;
  const unsigned short* p = Wstg + col * WST + kb;
  AB32 u;
  u.q[0] = *(const uint4*)(p);
  u.q[1] = *(const uint4*)(p + 16);
  return u.v;
}

// Boundary means: top_c/left_c from cell_states, top_h (both sweeps) / left_h from hidden.
__global__ void stlstm_prep(const float* __restrict__ hidden,
                            const float* __restrict__ cell,
                            const float* __restrict__ gts,
                            float* __restrict__ topc,  float* __restrict__ leftc,
                            float* __restrict__ toph0, float* __restrict__ toph1,
                            float* __restrict__ lefth) {
  int tid = blockIdx.x * blockDim.x + threadIdx.x;         // [0, B*NB*H)
  {
    int b = tid / (NB_ * H_);
    int rem = tid % (NB_ * H_);
    int nb = rem / H_, h = rem % H_;
    float sc = 0.f, sh = 0.f;
    for (int f = 0; f < F_; ++f) {
      size_t idx = ((size_t)(b * F_ + f) * NB_ + nb) * H_ + h;
      sc += cell[idx]; sh += hidden[idx];
    }
    topc [tid] = sc * (1.f / F_);
    toph0[tid] = sh * (1.f / F_);
    toph1[tid] = (sh + gts[(size_t)(b * NB_ + nb) * H_ + h]) * (1.f / (F_ + 1));
  }
  if (tid < B_ * F_ * H_) {
    int b = tid / (F_ * H_);
    int f = (tid / H_) % F_;
    int h = tid % H_;
    float sc = 0.f, sh = 0.f;
    for (int nb = 0; nb < NB_; ++nb) {
      size_t idx = ((size_t)(b * F_ + f) * NB_ + nb) * H_ + h;
      sc += cell[idx]; sh += hidden[idx];
    }
    leftc[tid] = sc * (1.f / NB_);
    lefth[tid] = sh * (1.f / NB_);
  }
}

// One block = one grid cell (f,nb) x one 64-column half. 8 waves, each wave owns
// (m = batch half, nloc = 16-col tile) and all 5 gate accumulators for that tile,
// so the gate combine is entirely in-register. Weight slabs stream HBM->LDS through
// a double buffer: stage (s+1) overlaps compute (s) on independent LOADcnt/DScnt.
// The 15-stage loop is fully unrolled so acc[g] indices are compile-time constants
// (keeps the accumulators in VGPRs -- no scratch spills around the WMMAs).
__global__ __launch_bounds__(256)
void stlstm_cell(const float* __restrict__ p,
                 const float* __restrict__ U,
                 const float* __restrict__ Wt,
                 const float* __restrict__ Ws,
                 const float* __restrict__ bias,
                 const float* __restrict__ topc, const float* __restrict__ leftc,
                 const float* __restrict__ toph, const float* __restrict__ lefth,
                 float* __restrict__ hst, float* __restrict__ cst,
                 int r, int fmin, int d) {
  const int f    = fmin + (int)blockIdx.x;
  const int nb   = d - f;
  const int cb   = (int)blockIdx.y * 64;       // output-column base
  const int tid  = threadIdx.x;
  const int lane = tid & 31;
  const int wv   = tid >> 5;
  const int m    = wv & 1;                     // batch-row tile (0..1)
  const int nloc = wv >> 1;                    // local 16-col tile (0..3)

  __shared__ unsigned short Ax[32 * ABST];     // x        as bf16, [b][k]
  __shared__ unsigned short At[32 * ABST];     // h_top    as bf16
  __shared__ unsigned short As[32 * ABST];     // h_left   as bf16
  __shared__ unsigned short Wstg[2][64 * WST]; // double-buffered weight tile, transposed [n][k]

  // ---- stage the three A operands (fp32 -> bf16) ----
  const float* xsrc = (r == 0) ? p : hst;      // sweep 0 reads p; sweep 1 reads previous h in-place
  for (int i = tid; i < (32 * 128) / 4; i += 256) {
    int flat = i * 4;
    int b = flat >> 7;
    int h = flat & 127;
    float4 vx = *(const float4*)(xsrc + ((size_t)(b * F_ + f) * NB_ + nb) * H_ + h);
    float4 vt = (f == 0)
        ? *(const float4*)(toph + ((size_t)b * NB_ + nb) * H_ + h)
        : *(const float4*)(hst + ((size_t)(b * F_ + (f - 1)) * NB_ + nb) * H_ + h);
    float4 vs = (nb == 0)
        ? *(const float4*)(lefth + ((size_t)(b * F_ + f)) * H_ + h)
        : *(const float4*)(hst + ((size_t)(b * F_ + f) * NB_ + (nb - 1)) * H_ + h);
    int o = b * ABST + h;
    *(ushort4*)&Ax[o] = make_ushort4(f2bf(vx.x), f2bf(vx.y), f2bf(vx.z), f2bf(vx.w));
    *(ushort4*)&At[o] = make_ushort4(f2bf(vt.x), f2bf(vt.y), f2bf(vt.z), f2bf(vt.w));
    *(ushort4*)&As[o] = make_ushort4(f2bf(vs.x), f2bf(vs.y), f2bf(vs.z), f2bf(vs.w));
  }

  const size_t cellw = (size_t)((r * F_ + f) * NB_ + nb) * (size_t)(G_ * H_ * H_);

  // coalesced fp32 load of one gate's 128x64 weight slab -> bf16, transposed into LDS
  #define STAGE_W(SRC, GATE, DST)                                              \
    {                                                                          \
      const float* Wg_ = ((SRC) == 0 ? U : ((SRC) == 1 ? Wt : Ws)) + cellw +   \
                         (size_t)(GATE) * H_ * H_;                             \
      unsigned short* dst_ = (DST);                                            \
      for (int i = tid; i < (64 * 128) / 4; i += 256) {                        \
        int flat = i * 4;                                                      \
        int kk = flat >> 6;                                                    \
        int n  = flat & 63;                                                    \
        float4 w = *(const float4*)(Wg_ + (size_t)kk * H_ + cb + n);           \
        dst_[(n + 0) * WST + kk] = f2bf(w.x);                                  \
        dst_[(n + 1) * WST + kk] = f2bf(w.y);                                  \
        dst_[(n + 2) * WST + kk] = f2bf(w.z);                                  \
        dst_[(n + 3) * WST + kk] = f2bf(w.w);                                  \
      }                                                                        \
    }

  STAGE_W(0, 0, Wstg[0]);
  __syncthreads();                              // A operands + Wstg[0] visible

  f32x8  acc[G_] = {};
  bf16x16 afr[4];

  const int NSTAGE = 3 * G_;                    // 15 (src, gate) stages
  #pragma unroll                                 // full unroll: sr/g/parity become constants
  for (int s = 0; s < NSTAGE; ++s) {
    const int sr = s / G_;
    const int g  = s % G_;
    // stage next weight slab into the other buffer (overlaps with WMMAs below)
    if (s + 1 < NSTAGE) {
      const int s2 = s + 1;
      STAGE_W(s2 / G_, s2 % G_, Wstg[s2 & 1]);
    }
    // A fragments are shared by all 5 gates of this src: load once per src group
    if (g == 0) {
      const unsigned short* Ab = (sr == 0) ? Ax : ((sr == 1) ? At : As);
      #pragma unroll
      for (int kc = 0; kc < 4; ++kc) afr[kc] = load_a_frag(Ab, m, kc, lane);
    }
    const unsigned short* Wb = Wstg[s & 1];
    #pragma unroll
    for (int kc = 0; kc < 4; ++kc) {            // K = 128 in four 32-wide chunks
      bf16x16 bfr = load_b_frag(Wb, nloc, kc, lane);
      acc[g] = __builtin_amdgcn_wmma_f32_16x16x32_bf16(
          false, afr[kc], false, bfr, (short)0, acc[g], false, false);
    }
    __syncthreads();                            // next buffer staged; this buffer free in 2 stages
  }
  #undef STAGE_W

  // ---- epilogue: bias + gates + cell/hidden update, fully in-register per wave ----
  const int kcol = cb + nloc * 16 + (lane & 15);
  const int lh   = (lane >> 4) & 1;
  float zb[G_];
  #pragma unroll
  for (int g = 0; g < G_; ++g)
    zb[g] = bias[((size_t)((r * F_ + f) * NB_ + nb) * G_ + g) * H_ + kcol];

  #pragma unroll
  for (int j = 0; j < 8; ++j) {                 // C/D layout: rows M = m*16 + lh*8 + j, col = kcol
    int b = m * 16 + lh * 8 + j;
    float i_ = sigm (acc[0][j] + zb[0]);
    float fs = sigm (acc[1][j] + zb[1]);
    float ft = sigm (acc[2][j] + zb[2]);
    float o_ = sigm (acc[3][j] + zb[3]);
    float cn = tanhf_(acc[4][j] + zb[4]);
    float cl = (nb == 0) ? leftc[((size_t)(b * F_ + f)) * H_ + kcol]
                         : cst[((size_t)(b * F_ + f) * NB_ + (nb - 1)) * H_ + kcol];
    float ct = (f == 0)  ? topc[((size_t)b * NB_ + nb) * H_ + kcol]
                         : cst[((size_t)(b * F_ + (f - 1)) * NB_ + nb) * H_ + kcol];
    float c2 = i_ * cn + fs * cl + ft * ct;
    float h2 = o_ * tanhf_(c2);
    size_t oo = ((size_t)(b * F_ + f) * NB_ + nb) * H_ + kcol;
    hst[oo] = h2;
    cst[oo] = c2;
  }
}

extern "C" void kernel_launch(void* const* d_in, const int* in_sizes, int n_in,
                              void* d_out, int out_size, void* d_ws, size_t ws_size,
                              hipStream_t stream) {
  (void)in_sizes; (void)n_in; (void)out_size; (void)ws_size;
  const float* hidden = (const float*)d_in[0];
  const float* cell   = (const float*)d_in[1];
  const float* gts    = (const float*)d_in[2];
  // d_in[3] = global_s_state: dead code for R==2 (the i>=2 branch never executes)
  const float* p    = (const float*)d_in[4];
  const float* U    = (const float*)d_in[5];
  const float* Wt   = (const float*)d_in[6];
  const float* Ws   = (const float*)d_in[7];
  const float* bias = (const float*)d_in[8];

  float* hst = (float*)d_out;                          // h state lives in d_out (in-place wavefront)
  float* cst = hst + (size_t)B_ * F_ * NB_ * H_;

  float* wsf   = (float*)d_ws;                         // 327,680 floats (~1.3 MB) of scratch
  float* topc  = wsf;
  float* leftc = topc  + (size_t)B_ * NB_ * H_;
  float* toph0 = leftc + (size_t)B_ * F_  * H_;
  float* toph1 = toph0 + (size_t)B_ * NB_ * H_;
  float* lefth = toph1 + (size_t)B_ * NB_ * H_;

  stlstm_prep<<<(B_ * NB_ * H_) / 256, 256, 0, stream>>>(
      hidden, cell, gts, topc, leftc, toph0, toph1, lefth);

  for (int r = 0; r < R_; ++r) {
    const float* toph = (r == 0) ? toph0 : toph1;
    for (int dd = 0; dd <= F_ + NB_ - 2; ++dd) {       // 29 wavefront diagonals
      int fmin = dd - (NB_ - 1); if (fmin < 0) fmin = 0;
      int fmax = (dd < F_ - 1) ? dd : F_ - 1;
      dim3 grid((unsigned)(fmax - fmin + 1), 2);
      stlstm_cell<<<grid, 256, 0, stream>>>(
          p, U, Wt, Ws, bias, topc, leftc, toph, lefth, hst, cst, r, fmin, dd);
    }
  }
}
